// DMN4_80444737454117
// MI455X (gfx1250) — matrix operand order repaired and checked by reading, not compiled, not run on hardware
//
#include <hip/hip_runtime.h>
#include <hip/hip_bf16.h>
#include <math.h>

// ---------------------------------------------------------------- constants
#define B_DIM 4
#define Q_DIM 75
#define C_DIM 640
#define HW 100
#define N_WAY 5
#define K_SHOT 5
#define S_DIM (N_WAY * K_SHOT)       // 25
#define MS 500                        // K_SHOT*HW support descriptors / class
#define MQ 100                        // query descriptors / query
#define NCOLS 2500                    // N_WAY*MS
#define NCOLS_PAD 2560                // 160 tiles of 16
#define NT_TILES 160
#define MT_TILES 7                    // 112 rows (100 real + pad)
#define KT_STEPS 20                   // 640 / 32
#define FRAG_ELEMS 512                // 32 lanes * 16 bf16
#define BQ_TOTAL (B_DIM * Q_DIM)      // 300

// workspace layout (bf16 stored as u16)
#define B_ELEMS ((size_t)B_DIM * NT_TILES * KT_STEPS * FRAG_ELEMS)       // 6,553,600
#define A_ELEMS_PER_BQ ((size_t)MT_TILES * KT_STEPS * FRAG_ELEMS)        // 71,680
#define A_ELEMS ((size_t)BQ_TOTAL * A_ELEMS_PER_BQ)                      // 21,504,000
#define B_BYTES (B_ELEMS * 2)
#define A_BYTES (A_ELEMS * 2)
#define LDS_A_BYTES (A_ELEMS_PER_BQ * 2)                                 // 143,360

typedef __attribute__((ext_vector_type(16))) __bf16 v16bf;
typedef __attribute__((ext_vector_type(8)))  float  v8f;

union V16U { uint4 u4[2]; v16bf v; };

// float -> bf16 round-to-nearest-even (matches HW RNE)
__device__ __forceinline__ unsigned short f2bf(float f) {
    unsigned int b = __float_as_uint(f);
    unsigned int r = b + 0x7FFFu + ((b >> 16) & 1u);
    return (unsigned short)(r >> 16);
}
// order-preserving float<->uint (max over uints == max over floats)
__device__ __forceinline__ unsigned int ford(float f) {
    unsigned int b = __float_as_uint(f);
    return (b & 0x80000000u) ? ~b : (b | 0x80000000u);
}
__device__ __forceinline__ float finv(unsigned int o) {
    unsigned int b = (o & 0x80000000u) ? (o ^ 0x80000000u) : ~o;
    return __uint_as_float(b);
}

// ---------------------------------------------------------------- kernels
__global__ void dmn4_zero_ws(unsigned int* p, long long nwords) {
    long long i = (long long)blockIdx.x * blockDim.x + threadIdx.x;
    long long stride = (long long)gridDim.x * blockDim.x;
    for (; i < nwords; i += stride) p[i] = 0u;
}

// Support: normalize each descriptor (b,si,p) over c, scatter into B-fragment
// layout [b][nt][kt][lane][e].  B 32x16 bf16 layout: col = lane%16,
// lanes 0-15 hold K 0-7,16-23, lanes 16-31 hold K 8-15,24-31, e = K%8 + 8*(K/16).
__global__ void dmn4_norm_support(const float* __restrict__ sup,
                                  unsigned short* __restrict__ Bf) {
    int b = blockIdx.x / S_DIM, si = blockIdx.x % S_DIM;
    const float* src = sup + (size_t)(b * S_DIM + si) * C_DIM * HW;
    __shared__ float rnorm[HW];
    int tid = threadIdx.x;
    if (tid < HW) {
        float ss = 0.f;
        for (int c = 0; c < C_DIM; ++c) {
            float x = src[c * HW + tid];
            ss += x * x;
        }
        rnorm[tid] = 1.0f / fmaxf(sqrtf(ss), 1e-12f);
    }
    __syncthreads();
    for (int idx = tid; idx < C_DIM * HW; idx += blockDim.x) {
        int c = idx / HW, p = idx % HW;
        unsigned short h = f2bf(src[c * HW + p] * rnorm[p]);
        int j = si * HW + p;              // global support column 0..2499
        int nt = j >> 4, nl = j & 15;
        int kt = c >> 5, kl = c & 31;
        int lane = nl + ((kl & 8) ? 16 : 0);
        int e = (kl & 7) + ((kl & 16) ? 8 : 0);
        Bf[((size_t)(b * NT_TILES + nt) * KT_STEPS + kt) * FRAG_ELEMS + lane * 16 + e] = h;
    }
}

// Query: normalize (b,qi,p) over c, scatter into A-fragment layout
// [bq][kt][mt][lane][e].  A 16x32 bf16 layout: row = lane%16, lanes 16-31 are
// K-offset +8 within each 16-half; VGPR0-3 = K 0..15, VGPR4-7 = K 16..31.
__global__ void dmn4_norm_query(const float* __restrict__ qry,
                                unsigned short* __restrict__ Af) {
    int bq = blockIdx.x;
    const float* src = qry + (size_t)bq * C_DIM * HW;
    __shared__ float rnorm[HW];
    int tid = threadIdx.x;
    if (tid < HW) {
        float ss = 0.f;
        for (int c = 0; c < C_DIM; ++c) {
            float x = src[c * HW + tid];
            ss += x * x;
        }
        rnorm[tid] = 1.0f / fmaxf(sqrtf(ss), 1e-12f);
    }
    __syncthreads();
    unsigned short* dst = Af + (size_t)bq * A_ELEMS_PER_BQ;
    for (int idx = tid; idx < C_DIM * HW; idx += blockDim.x) {
        int c = idx / HW, p = idx % HW;      // m = p
        unsigned short h = f2bf(src[c * HW + p] * rnorm[p]);
        int mt = p >> 4, ml = p & 15;
        int kt = c >> 5, kl = c & 31;
        int lane = ml + ((kl & 8) ? 16 : 0);
        int v = ((kl & 16) ? 4 : 0) + ((kl & 7) >> 1);
        int e = 2 * v + (kl & 1);
        dst[((size_t)kt * MT_TILES + mt) * FRAG_ELEMS + lane * 16 + e] = h;
    }
}

// Fused GEMM + reductions: one workgroup per (b,q).
__global__ void __launch_bounds__(256)
dmn4_gemm(const unsigned short* __restrict__ Af,
          const unsigned short* __restrict__ Bf,
          const int* __restrict__ qy,
          float* __restrict__ lossws) {
    extern __shared__ unsigned short ldsA[];           // 143,360 B
    __shared__ unsigned int cmax_u[112 * 8];           // ordered-float class max
    __shared__ unsigned long long nkey[112];           // (sim_ord<<32)|~j
    __shared__ float cmf[112 * 8];
    __shared__ float diffv[112];
    __shared__ int   nearestj[112];
    __shared__ int   maskv[112];
    __shared__ float qvv[N_WAY];

    int bq = blockIdx.x;
    int b = bq / Q_DIM;
    int tid = threadIdx.x;
    int wave = tid >> 5, lane = tid & 31;

    // stage this (b,q)'s A fragments into LDS (coalesced b128 copies)
    const uint4* Ag4 = (const uint4*)(Af + (size_t)bq * A_ELEMS_PER_BQ);
    uint4* As4 = (uint4*)ldsA;
    for (int i = tid; i < (int)(LDS_A_BYTES / 16); i += 256) As4[i] = Ag4[i];
    for (int i = tid; i < 112 * 8; i += 256) cmax_u[i] = 0u;
    for (int i = tid; i < 112; i += 256) nkey[i] = 0ull;
    __syncthreads();

    // each wave: 20 N-tiles, accumulate 7 M-tiles x f32 over K=640
    const v8f zero8 = {0.f, 0.f, 0.f, 0.f, 0.f, 0.f, 0.f, 0.f};
    for (int nt = wave; nt < NT_TILES; nt += 8) {
        v8f acc[MT_TILES];
#pragma unroll
        for (int mt = 0; mt < MT_TILES; ++mt) acc[mt] = zero8;

        const uint4* Bg4 = (const uint4*)(Bf + (size_t)(b * NT_TILES + nt) * KT_STEPS * FRAG_ELEMS);
        for (int kt = 0; kt < KT_STEPS; ++kt) {
            if (kt + 1 < KT_STEPS)
                __builtin_prefetch(Bg4 + (kt + 1) * 64 + lane * 2, 0, 1);
            V16U bv;
            bv.u4[0] = Bg4[kt * 64 + lane * 2 + 0];
            bv.u4[1] = Bg4[kt * 64 + lane * 2 + 1];
            // hoist all 7 A fragments first so the DS clause can issue once
            // and WMMAs retire against staggered (non-zero) dscnt waits.
            V16U av[MT_TILES];
#pragma unroll
            for (int mt = 0; mt < MT_TILES; ++mt) {
                int base4 = (kt * MT_TILES + mt) * 64 + lane * 2;
                av[mt].u4[0] = As4[base4];
                av[mt].u4[1] = As4[base4 + 1];
            }
#pragma unroll
            for (int mt = 0; mt < MT_TILES; ++mt) {
                acc[mt] = __builtin_amdgcn_wmma_f32_16x16x32_bf16(
                    false, av[mt].v, false, bv.v, (short)0, acc[mt], false, false);
            }
        }

        // fold this N-tile into LDS reductions.  C layout: col = lane%16,
        // rows = (lane<16 ? 0..7 : 8..15) + 16*mt across the 8 acc regs.
        int ncol = nt * 16 + (lane & 15);
        if (ncol < NCOLS) {
            int cls = ncol / MS;
            unsigned int invj = 0xFFFFFFFFu - (unsigned int)ncol;
            int mbase = (lane < 16) ? 0 : 8;
#pragma unroll
            for (int mt = 0; mt < MT_TILES; ++mt) {
#pragma unroll
                for (int v = 0; v < 8; ++v) {
                    int m = mt * 16 + mbase + v;
                    if (m < MQ) {
                        unsigned int os = ford(acc[mt][v]);
                        atomicMax(&cmax_u[m * 8 + cls], os);
                        atomicMax(&nkey[m], ((unsigned long long)os << 32) | invj);
                    }
                }
            }
        }
    }
    __syncthreads();

    // per-m: decode class max, top-2 gap, nearest column
    if (tid < MQ) {
        float t1 = -1e30f, t2 = -1e30f;
        for (int n = 0; n < N_WAY; ++n) {
            float v = finv(cmax_u[tid * 8 + n]);
            cmf[tid * 8 + n] = v;
            if (v > t1) { t2 = t1; t1 = v; }
            else if (v > t2) { t2 = v; }
        }
        diffv[tid] = t1 - t2;
        nearestj[tid] = (int)(0xFFFFFFFFu - (unsigned int)(nkey[tid] & 0xFFFFFFFFull));
    }
    __syncthreads();

    // mutual-NN mask: argmax over rows of one-hot(nearest)*diff, ties->lowest m,
    // all-zero column -> index 0 (jnp.argmax semantics).
    if (tid < MQ) {
        int j = nearestj[tid];
        float maxd = -1.f;
        int win = 0;
        for (int mp = 0; mp < MQ; ++mp) {
            float v = (nearestj[mp] == j) ? diffv[mp] : 0.f;
            if (v > maxd) { maxd = v; win = mp; }
        }
        maskv[tid] = (win == tid) ? 1 : 0;
    }
    __syncthreads();

    if (tid < N_WAY) {
        float s = 0.f;
        for (int m = 0; m < MQ; ++m)
            if (maskv[m]) s += cmf[m * 8 + tid];
        qvv[tid] = s;
    }
    __syncthreads();

    if (tid == 0) {
        float l[N_WAY], mx = -1e30f;
        for (int n = 0; n < N_WAY; ++n) {
            l[n] = qvv[n] * 5.0f;          // /TEMPERATURE (0.2)
            mx = fmaxf(mx, l[n]);
        }
        float se = 0.f;
        for (int n = 0; n < N_WAY; ++n) se += expf(l[n] - mx);
        float lse = logf(se) + mx;
        int label = qy[bq];
        lossws[bq] = lse - l[label];       // -log_softmax[label]
    }
}

__global__ void dmn4_reduce(const float* __restrict__ lossws, float* __restrict__ out) {
    __shared__ float buf[256];
    int tid = threadIdx.x;
    float s = 0.f;
    for (int i = tid; i < BQ_TOTAL; i += 256) s += lossws[i];
    buf[tid] = s;
    __syncthreads();
    for (int off = 128; off > 0; off >>= 1) {
        if (tid < off) buf[tid] += buf[tid + off];
        __syncthreads();
    }
    if (tid == 0) out[0] = buf[0] / (float)BQ_TOTAL;
}

// ---------------------------------------------------------------- launch
extern "C" void kernel_launch(void* const* d_in, const int* in_sizes, int n_in,
                              void* d_out, int out_size, void* d_ws, size_t ws_size,
                              hipStream_t stream) {
    const float* support = (const float*)d_in[0];
    // d_in[1] (support_y) is unused by the reference
    const float* query = (const float*)d_in[2];
    const int* query_y = (const int*)d_in[3];

    unsigned short* Bf = (unsigned short*)d_ws;
    unsigned short* Af = (unsigned short*)((char*)d_ws + B_BYTES);
    float* lossws = (float*)((char*)d_ws + B_BYTES + A_BYTES);
    float* out = (float*)d_out;

    long long nwords = (long long)((B_BYTES + A_BYTES) / 4);
    dmn4_zero_ws<<<2048, 256, 0, stream>>>((unsigned int*)d_ws, nwords);
    dmn4_norm_support<<<B_DIM * S_DIM, 256, 0, stream>>>(support, Bf);
    dmn4_norm_query<<<BQ_TOTAL, 256, 0, stream>>>(query, Af);
    dmn4_gemm<<<BQ_TOTAL, 256, LDS_A_BYTES, stream>>>(Af, Bf, query_y, lossws);
    dmn4_reduce<<<1, 256, 0, stream>>>(lossws, out);
}
